// VisionAttention_48206712930376
// MI455X (gfx1250) — matrix operand order repaired and checked by reading, not compiled, not run on hardware
//
#include <hip/hip_runtime.h>
#include <hip/hip_bf16.h>

typedef _Float16 v16h __attribute__((ext_vector_type(16)));
typedef _Float16 v2h  __attribute__((ext_vector_type(2)));
typedef float    v8f  __attribute__((ext_vector_type(8)));
typedef float    v2f  __attribute__((ext_vector_type(2)));

#define D_MODEL 512
#define NTOK    1024
#define NBATCH  16
#define MTOT    (NBATCH * NTOK)   /* 16384 token rows total */

#define WMMA_F16(a, b, c) \
  __builtin_amdgcn_wmma_f32_16x16x32_f16(false, (a), false, (b), (short)0, (c), false, false)

// ---------------- WMMA fragment loaders (wave32 layouts per CDNA5 ISA 7.12.2) ----

// A: 16x32 f16 tile from row-major [row][k], stride ld. lane&15 = M row.
__device__ __forceinline__ v16h load_a16(const _Float16* __restrict__ base, int ld, int lane) {
  const int m = lane & 15, h = lane >> 4;
  const _Float16* row = base + m * ld + (h << 3);   // +8*half on K
  v16h a;
#pragma unroll
  for (int i = 0; i < 8; ++i) {
    const int k = ((i & 4) << 2) + ((i & 3) << 1);  // (i>=4?16:0) + 2*(i&3)
    v2h p = *(const v2h*)(row + k);
    a[2*i] = p[0]; a[2*i+1] = p[1];
  }
  return a;
}

// A: 16x32 tile, source is f32 row-major (attention probs) -> convert to f16.
__device__ __forceinline__ v16h load_a32(const float* __restrict__ base, int ld, int lane) {
  const int m = lane & 15, h = lane >> 4;
  const float* row = base + m * ld + (h << 3);
  v16h a;
#pragma unroll
  for (int i = 0; i < 8; ++i) {
    const int k = ((i & 4) << 2) + ((i & 3) << 1);
    v2f p = *(const v2f*)(row + k);
    a[2*i] = (_Float16)p[0]; a[2*i+1] = (_Float16)p[1];
  }
  return a;
}

// B: 32x16 f16 tile B[k][n], sourced from row-major [n][k] storage, stride ld.
__device__ __forceinline__ v16h load_bt16(const _Float16* __restrict__ base, int ld, int lane) {
  const int n = lane & 15, h = lane >> 4;
  const _Float16* row = base + n * ld + (h << 3);
  v16h b;
#pragma unroll
  for (int j = 0; j < 8; ++j) {
    v2h p = *(const v2h*)(row + ((j & 4) << 2) + ((j & 3) << 1));
    b[2*j] = p[0]; b[2*j+1] = p[1];
  }
  return b;
}

// ---------------- Ping-pong software-pipelined 32x64 GEMM core -----------------
// Two A fragments (M=32) x four B fragments (N=64); 8 WMMAs per K-step.
// K-loop unrolled by 2 with two fragment buffers alternating roles, so there
// is no register rotation (no v_mov copies) between pipeline stages.

struct Frags { v16h a0, a1, b0, b1, b2, b3; };

template <typename ALoad>
__device__ __forceinline__ void load_frags(Frags& f, ALoad& la,
                                           const _Float16* __restrict__ bbase,
                                           int ldb, int k0, int lane) {
  f.a0 = la(0, k0);
  f.a1 = la(1, k0);
  f.b0 = load_bt16(bbase + k0,                    ldb, lane);
  f.b1 = load_bt16(bbase + (size_t)16 * ldb + k0, ldb, lane);
  f.b2 = load_bt16(bbase + (size_t)32 * ldb + k0, ldb, lane);
  f.b3 = load_bt16(bbase + (size_t)48 * ldb + k0, ldb, lane);
}

__device__ __forceinline__ void mma_frags(const Frags& f, v8f acc[8]) {
  acc[0] = WMMA_F16(f.a0, f.b0, acc[0]);
  acc[1] = WMMA_F16(f.a0, f.b1, acc[1]);
  acc[2] = WMMA_F16(f.a0, f.b2, acc[2]);
  acc[3] = WMMA_F16(f.a0, f.b3, acc[3]);
  acc[4] = WMMA_F16(f.a1, f.b0, acc[4]);
  acc[5] = WMMA_F16(f.a1, f.b1, acc[5]);
  acc[6] = WMMA_F16(f.a1, f.b2, acc[6]);
  acc[7] = WMMA_F16(f.a1, f.b3, acc[7]);
}

template <typename ALoad>
__device__ __forceinline__ void wmma_gemm_32x64(ALoad la, const _Float16* __restrict__ bbase,
                                                int ldb, int K, int lane, v8f acc[8]) {
  Frags f0, f1;
  load_frags(f0, la, bbase, ldb, 0, lane);
  // K/32 is even for all call sites (16 or 32 K-steps).
  for (int k0 = 32; k0 < K - 32; k0 += 64) {
    load_frags(f1, la, bbase, ldb, k0, lane);
    __builtin_prefetch((const void*)(bbase + k0 + 96), 0, 1);  // global_prefetch_b8
    mma_frags(f0, acc);
    load_frags(f0, la, bbase, ldb, k0 + 32, lane);
    mma_frags(f1, acc);
  }
  load_frags(f1, la, bbase, ldb, K - 32, lane);
  mma_frags(f0, acc);
  mma_frags(f1, acc);
}

// ---------------- Stage 0: pack x (transpose [B,C,N] f32 -> [B,N,C] f16) -------

__global__ __launch_bounds__(256) void pack_x_kernel(const float* __restrict__ x,
                                                     _Float16* __restrict__ x16) {
  __shared__ float tile[32][33];
  const int blk = blockIdx.x;          // 16 * 16 * 32 = 8192 blocks
  const int nt  = blk & 31;
  const int ct  = (blk >> 5) & 15;
  const int b   = blk >> 9;
  const int tx  = threadIdx.x & 31;
  const int ty4 = threadIdx.x >> 5;
  const float* xb = x + ((size_t)b * D_MODEL + ct * 32) * NTOK + nt * 32;
#pragma unroll
  for (int r = 0; r < 4; ++r) {
    const int c = ty4 * 4 + r;
    tile[c][tx] = xb[(size_t)c * NTOK + tx];
  }
  __syncthreads();
  _Float16* ob = x16 + ((size_t)b * NTOK + nt * 32) * D_MODEL + ct * 32;
#pragma unroll
  for (int r = 0; r < 4; ++r) {
    const int n = ty4 * 4 + r;
    ob[(size_t)n * D_MODEL + tx] = (_Float16)tile[tx][n];
  }
}

// ---------------- Stage 0b: pack weights to f16 (layout [o][c] preserved) ------

__global__ __launch_bounds__(256) void pack_w_kernel(const float* __restrict__ wq,
                                                     const float* __restrict__ wk,
                                                     const float* __restrict__ wv,
                                                     const float* __restrict__ wo,
                                                     _Float16* __restrict__ w16) {
  const int idx = blockIdx.x * 256 + threadIdx.x;       // 4 * 262144
  const int mat = idx >> 18;
  const int off = idx & 262143;
  const float* src = (mat == 0) ? wq : (mat == 1) ? wk : (mat == 2) ? wv : wo;
  w16[idx] = (_Float16)src[off];
}

// ---------------- Stage 1: projections. One wave -> 32x64 tile, K=512 ----------

__global__ __launch_bounds__(256) void proj_kernel(const _Float16* __restrict__ a16,
                                                   const _Float16* __restrict__ w16,
                                                   const float* __restrict__ bias,
                                                   _Float16* __restrict__ out,
                                                   float scale, int vmode) {
  const int lane = threadIdx.x & 31;
  const int g = blockIdx.x * 8 + (threadIdx.x >> 5);    // 4096 waves
  const int mtile = g >> 3;                             // 0..511 (32-row tiles)
  const int n0 = (g & 7) * 64;
  const _Float16* abase = a16 + (size_t)mtile * 32 * D_MODEL;
  v8f acc[8] = {};
  wmma_gemm_32x64(
      [&](int mf, int k0) { return load_a16(abase + (size_t)mf * 16 * D_MODEL + k0, D_MODEL, lane); },
      w16 + (size_t)n0 * D_MODEL, D_MODEL, D_MODEL, lane, acc);
  const int h = lane >> 4;
#pragma unroll
  for (int mf = 0; mf < 2; ++mf) {
#pragma unroll
    for (int t = 0; t < 4; ++t) {
      const int o = n0 + t * 16 + (lane & 15);
      const float bv = bias[o];
#pragma unroll
      for (int r = 0; r < 8; ++r) {
        const int m = mtile * 32 + mf * 16 + r + (h << 3);
        const float val = (acc[mf * 4 + t][r] + bv) * scale;
        if (!vmode) {
          out[(size_t)m * D_MODEL + o] = (_Float16)val;               // [m][c]
        } else {
          const int bb = m >> 10, n = m & 1023;                       // v transposed
          out[((size_t)bb * D_MODEL + o) * NTOK + n] = (_Float16)val; // [b][c][n]
        }
      }
    }
  }
}

// ---------------- Stage 2: scores = q_scaled @ k^T  (per batch 1024x1024, K=512)

__global__ __launch_bounds__(256) void scores_kernel(const _Float16* __restrict__ q16,
                                                     const _Float16* __restrict__ k16,
                                                     float* __restrict__ attn) {
  const int lane = threadIdx.x & 31;
  const int g = blockIdx.x * 8 + (threadIdx.x >> 5);    // 8192 waves
  const int b = g >> 9;
  const int rem = g & 511;
  const int mtile = rem >> 4;                           // 0..31 (32 query rows)
  const int n0 = (rem & 15) * 64;                       // key-column base
  const _Float16* abase = q16 + ((size_t)b * NTOK + mtile * 32) * D_MODEL;
  const _Float16* bbase = k16 + ((size_t)b * NTOK + n0) * D_MODEL;
  v8f acc[8] = {};
  wmma_gemm_32x64(
      [&](int mf, int k0) { return load_a16(abase + (size_t)mf * 16 * D_MODEL + k0, D_MODEL, lane); },
      bbase, D_MODEL, D_MODEL, lane, acc);
  const int h = lane >> 4;
  float* sb = attn + (size_t)b * NTOK * NTOK;
#pragma unroll
  for (int mf = 0; mf < 2; ++mf) {
#pragma unroll
    for (int t = 0; t < 4; ++t) {
      const int col = n0 + t * 16 + (lane & 15);
#pragma unroll
      for (int r = 0; r < 8; ++r) {
        const int row = mtile * 32 + mf * 16 + r + (h << 3);
        sb[(size_t)row * NTOK + col] = acc[mf * 4 + t][r];
      }
    }
  }
}

// ---------------- Stage 3: row softmax in place (1024 per row) -----------------

__global__ __launch_bounds__(256) void softmax_kernel(float* __restrict__ attn) {
  __shared__ float red[256];
  float* p = attn + (size_t)blockIdx.x * NTOK;          // 16384 rows
  const int t = threadIdx.x;
  float v[4];
  float mx = -3.4e38f;
#pragma unroll
  for (int i = 0; i < 4; ++i) { v[i] = p[t + 256 * i]; mx = fmaxf(mx, v[i]); }
  red[t] = mx; __syncthreads();
  for (int s = 128; s > 0; s >>= 1) {
    if (t < s) red[t] = fmaxf(red[t], red[t + s]);
    __syncthreads();
  }
  mx = red[0]; __syncthreads();
  float sum = 0.f;
#pragma unroll
  for (int i = 0; i < 4; ++i) { v[i] = __expf(v[i] - mx); sum += v[i]; }
  red[t] = sum; __syncthreads();
  for (int s = 128; s > 0; s >>= 1) {
    if (t < s) red[t] += red[t + s];
    __syncthreads();
  }
  const float inv = 1.0f / red[0];
#pragma unroll
  for (int i = 0; i < 4; ++i) p[t + 256 * i] = v[i] * inv;
}

// ---------------- Stage 4: out = attn @ v   (per batch 1024x512, K=1024) -------

__global__ __launch_bounds__(256) void attnv_kernel(const float* __restrict__ attn,
                                                    const _Float16* __restrict__ vt,
                                                    _Float16* __restrict__ out16) {
  const int lane = threadIdx.x & 31;
  const int g = blockIdx.x * 8 + (threadIdx.x >> 5);    // 4096 waves
  const int b = g >> 8;
  const int rem = g & 255;
  const int mtile = rem >> 3;                           // 0..31
  const int c0 = (rem & 7) * 64;
  const float* abase = attn + ((size_t)b * NTOK + mtile * 32) * NTOK;
  const _Float16* bbase = vt + ((size_t)b * D_MODEL + c0) * NTOK;
  v8f acc[8] = {};
  wmma_gemm_32x64(
      [&](int mf, int k0) { return load_a32(abase + (size_t)mf * 16 * NTOK + k0, NTOK, lane); },
      bbase, NTOK, NTOK, lane, acc);
  const int h = lane >> 4;
#pragma unroll
  for (int mf = 0; mf < 2; ++mf) {
#pragma unroll
    for (int t = 0; t < 4; ++t) {
      const int c = c0 + t * 16 + (lane & 15);
#pragma unroll
      for (int r = 0; r < 8; ++r) {
        const int n = mtile * 32 + mf * 16 + r + (h << 3);
        out16[((size_t)b * NTOK + n) * D_MODEL + c] = (_Float16)acc[mf * 4 + t][r];
      }
    }
  }
}

// ---------------- Stage 5: y = x + out @ Wo^T + bo  ([B,C,H,W] layout) ---------

__global__ __launch_bounds__(256) void outproj_kernel(const _Float16* __restrict__ o16,
                                                      const _Float16* __restrict__ wo16,
                                                      const float* __restrict__ bo,
                                                      const float* __restrict__ x,
                                                      float* __restrict__ y) {
  const int lane = threadIdx.x & 31;
  const int g = blockIdx.x * 8 + (threadIdx.x >> 5);    // 4096 waves
  const int mtile = g >> 3;                             // 0..511
  const int n0 = (g & 7) * 64;
  const _Float16* abase = o16 + (size_t)mtile * 32 * D_MODEL;
  v8f acc[8] = {};
  wmma_gemm_32x64(
      [&](int mf, int k0) { return load_a16(abase + (size_t)mf * 16 * D_MODEL + k0, D_MODEL, lane); },
      wo16 + (size_t)n0 * D_MODEL, D_MODEL, D_MODEL, lane, acc);
  const int h = lane >> 4;
#pragma unroll
  for (int mf = 0; mf < 2; ++mf) {
#pragma unroll
    for (int t = 0; t < 4; ++t) {
      const int o = n0 + t * 16 + (lane & 15);
      const float bv = bo[o];
#pragma unroll
      for (int r = 0; r < 8; ++r) {
        const int m = mtile * 32 + mf * 16 + r + (h << 3);
        const int bat = m >> 10, n = m & 1023;
        const size_t addr = ((size_t)bat * D_MODEL + o) * NTOK + n;   // [b][o][h][w]
        y[addr] = x[addr] + acc[mf * 4 + t][r] + bv;
      }
    }
  }
}

// ---------------- Host-side orchestration --------------------------------------

extern "C" void kernel_launch(void* const* d_in, const int* in_sizes, int n_in,
                              void* d_out, int out_size, void* d_ws, size_t ws_size,
                              hipStream_t stream) {
  const float* x  = (const float*)d_in[0];
  const float* Wq = (const float*)d_in[1];
  const float* bq = (const float*)d_in[2];
  const float* Wk = (const float*)d_in[3];
  const float* bk = (const float*)d_in[4];
  const float* Wv = (const float*)d_in[5];
  const float* bv = (const float*)d_in[6];
  const float* Wo = (const float*)d_in[7];
  const float* bo = (const float*)d_in[8];

  float* y_out = (float*)d_out;                                     // [16,512,32,32]
  float* attn  = (float*)d_out + (size_t)NBATCH * D_MODEL * NTOK;   // [16,1024,1024]

  // Workspace layout (f16 stages), ~82 MB total.
  char* ws = (char*)d_ws;
  const size_t SZ_TOK = (size_t)MTOT * D_MODEL * sizeof(_Float16);  // 16 MB
  _Float16* x16  = (_Float16*)(ws);
  _Float16* w16  = (_Float16*)(ws + SZ_TOK);                        // 4 matrices, 2 MB
  _Float16* q16  = (_Float16*)(ws + SZ_TOK + (size_t)4 * D_MODEL * D_MODEL * sizeof(_Float16));
  _Float16* k16  = q16 + (size_t)MTOT * D_MODEL;
  _Float16* vt16 = k16 + (size_t)MTOT * D_MODEL;
  _Float16* o16  = vt16 + (size_t)MTOT * D_MODEL;

  _Float16* wq16 = w16;
  _Float16* wk16 = w16 + (size_t)1 * D_MODEL * D_MODEL;
  _Float16* wv16 = w16 + (size_t)2 * D_MODEL * D_MODEL;
  _Float16* wo16 = w16 + (size_t)3 * D_MODEL * D_MODEL;

  const float inv_temp = 0.044194173824159216f;  // 1/sqrt(512)

  pack_x_kernel<<<8192, 256, 0, stream>>>(x, x16);
  pack_w_kernel<<<4096, 256, 0, stream>>>(Wq, Wk, Wv, Wo, w16);

  proj_kernel<<<512, 256, 0, stream>>>(x16, wq16, bq, q16, inv_temp, 0); // q/temp folded
  proj_kernel<<<512, 256, 0, stream>>>(x16, wk16, bk, k16, 1.0f, 0);
  proj_kernel<<<512, 256, 0, stream>>>(x16, wv16, bv, vt16, 1.0f, 1);    // transposed

  scores_kernel<<<1024, 256, 0, stream>>>(q16, k16, attn);
  softmax_kernel<<<16384, 256, 0, stream>>>(attn);
  attnv_kernel<<<512, 256, 0, stream>>>(attn, vt16, o16);
  outproj_kernel<<<512, 256, 0, stream>>>(o16, wo16, bo, x, y_out);
}
